// ScaleDotProductAttention_39651138077522
// MI455X (gfx1250) — matrix operand order repaired and checked by reading, not compiled, not run on hardware
//
#include <hip/hip_runtime.h>
#include <hip/hip_bf16.h>

typedef __attribute__((ext_vector_type(16))) __bf16 v16bf;
typedef __attribute__((ext_vector_type(8)))  float  v8f;

#define SEQ 2048
#define DIM 128
#define SSTRIDE 2052   // 16x2048 score rows padded by 4 floats to stagger LDS banks

// Packed fragment arrays in d_ws (bf16 elements):
//   K pack: [b:32][j:128][c:4][lane:32][sel:2][elem:16]  -> 16,777,216 elems
//   V pack: [b:32][kc:64][dt:8][lane:32][sel:2][elem:16] -> 16,777,216 elems
#define KP_ELEMS (32u * 128u * 4u * 32u * 32u)
#define VP_ELEMS (32u * 64u * 8u * 32u * 32u)

__device__ __forceinline__ v8f wmma_bf16(v16bf a, v16bf b, v8f c) {
  return __builtin_amdgcn_wmma_f32_16x16x32_bf16(
      /*neg_a=*/false, a, /*neg_b=*/false, b,
      /*c_mod=*/(short)0, c, /*reuse_a=*/false, /*reuse_b=*/false);
}

__device__ __forceinline__ void split_bf16(float x, __bf16& h, __bf16& l) {
  h = (__bf16)x;
  l = (__bf16)(x - (float)h);
}

// ---- Prep: pack K into B-fragment layout (hi/lo bf16), one thread per (frag,lane) ----
__global__ void pack_k_kernel(const float* __restrict__ kg, __bf16* __restrict__ kp) {
  const unsigned idx = blockIdx.x * blockDim.x + threadIdx.x;  // 2^19 threads
  const int L  = idx & 31;
  const int c  = (idx >> 5) & 3;
  const int j  = (idx >> 7) & 127;
  const int b  = idx >> 14;
  const int lm = L & 15, hg = L >> 4;

  const float* src = kg + (size_t)b * SEQ * DIM + (size_t)(j * 16 + lm) * DIM + c * 32 + hg * 16;
  v16bf hi, lo;
#pragma unroll
  for (int i = 0; i < 16; ++i) {
    __bf16 h, l; split_bf16(src[i], h, l);
    hi[i] = h; lo[i] = l;
  }
  v16bf* dst = (v16bf*)kp + (size_t)idx * 2;
  dst[0] = hi; dst[1] = lo;
}

// ---- Prep: pack V into B-fragment layout (hi/lo bf16) ----
__global__ void pack_v_kernel(const float* __restrict__ vg, __bf16* __restrict__ vp) {
  const unsigned idx = blockIdx.x * blockDim.x + threadIdx.x;  // 2^19 threads
  const int L  = idx & 31;
  const int dt = (idx >> 5) & 7;
  const int kc = (idx >> 8) & 63;
  const int b  = idx >> 14;
  const int lm = L & 15, hg = L >> 4;

  const float* src = vg + (size_t)b * SEQ * DIM + (size_t)(kc * 32 + hg * 16) * DIM + dt * 16 + lm;
  v16bf hi, lo;
#pragma unroll
  for (int i = 0; i < 16; ++i) {
    __bf16 h, l; split_bf16(src[(size_t)i * DIM], h, l);
    hi[i] = h; lo[i] = l;
  }
  v16bf* dst = (v16bf*)vp + (size_t)idx * 2;
  dst[0] = hi; dst[1] = lo;
}

template <bool PACKED>
__global__ void sdpa_kernel(const float* __restrict__ qg,
                            const float* __restrict__ kg,
                            const float* __restrict__ vg,
                            const __bf16* __restrict__ kp,
                            const __bf16* __restrict__ vp,
                            float* __restrict__ ctx,
                            float* __restrict__ attn) {
  extern __shared__ float smem[];
  float* sS      = smem;                   // [16][SSTRIDE] scores -> probs
  float* sRed    = smem + 16 * SSTRIDE;    // [128]
  float* sRowMax = sRed + 128;             // [16]
  float* sRowInv = sRowMax + 16;           // [16]

  const int qt = blockIdx.x;               // q tile (16 rows)
  const int b  = blockIdx.y;               // batch
  const int t  = threadIdx.x;              // 0..127
  const int L  = t & 31;
  const int w  = t >> 5;                   // wave 0..3
  const int lm = L & 15;
  const int hg = L >> 4;

  const float scale = 0.08838834764831845f; // 1/sqrt(128)

  const size_t batchBase = (size_t)b * SEQ * DIM;
  const float* Q = qg + batchBase + (size_t)qt * 16 * DIM;
  const float* K = kg + batchBase;
  const float* V = vg + batchBase;

  // ---- Q A-fragments (scaled, hi/lo split) — built once per block, cheap ----
  v16bf qa_h[4], qa_l[4];
#pragma unroll
  for (int c = 0; c < 4; ++c) {
    v16bf ah, al;
#pragma unroll
    for (int i = 0; i < 16; ++i) {
      const int Kd = i + (hg ? 16 : 0);
      float x = Q[lm * DIM + c * 32 + Kd] * scale;
      __bf16 h, l; split_bf16(x, h, l);
      ah[i] = h; al[i] = l;
    }
    qa_h[c] = ah; qa_l[c] = al;
  }

  // ---- Pass 1: S = (Q*scale) K^T (bf16x3); waves stripe key tiles ----
  const v16bf* kpv = (const v16bf*)kp;
  for (int j = w; j < SEQ / 16; j += 4) {
    v8f acc = {};
#pragma unroll
    for (int c = 0; c < 4; ++c) {
      v16bf bh, bl;
      if (PACKED) {
        const size_t base = ((size_t)(((b * 128 + j) * 4 + c) * 32 + L)) * 2;
        bh = kpv[base]; bl = kpv[base + 1];
      } else {
#pragma unroll
        for (int i = 0; i < 16; ++i) {
          const int Kd = i + (hg ? 16 : 0);
          float x = K[(j * 16 + lm) * DIM + c * 32 + Kd];
          __bf16 h, l; split_bf16(x, h, l);
          bh[i] = h; bl[i] = l;
        }
      }
      acc = wmma_bf16(qa_h[c], bh, acc);
      acc = wmma_bf16(qa_h[c], bl, acc);
      acc = wmma_bf16(qa_l[c], bh, acc);
    }
#pragma unroll
    for (int r2 = 0; r2 < 8; ++r2) {
      const int row = r2 + (hg ? 8 : 0);
      sS[row * SSTRIDE + j * 16 + lm] = acc[r2];
    }
  }
  __syncthreads();

  // ---- Softmax over full rows ----
  const int r   = t >> 3;
  const int sub = t & 7;
  float mx = -__builtin_inff();
  for (int kx = 0; kx < SEQ / 8; ++kx)
    mx = fmaxf(mx, sS[r * SSTRIDE + sub + kx * 8]);
  sRed[t] = mx;
  __syncthreads();
  if (t < 16) {
    float m = -__builtin_inff();
#pragma unroll
    for (int i = 0; i < 8; ++i) m = fmaxf(m, sRed[t * 8 + i]);
    sRowMax[t] = m;
  }
  __syncthreads();

  const float rm = sRowMax[r];
  float ssum = 0.f;
  for (int kx = 0; kx < SEQ / 8; ++kx) {
    const int col = sub + kx * 8;
    float p = __expf(sS[r * SSTRIDE + col] - rm);
    sS[r * SSTRIDE + col] = p;
    ssum += p;
  }
  sRed[t] = ssum;
  __syncthreads();
  if (t < 16) {
    float sm = 0.f;
#pragma unroll
    for (int i = 0; i < 8; ++i) sm += sRed[t * 8 + i];
    sRowInv[t] = 1.f / sm;
  }
  __syncthreads();

  // ---- Normalize in LDS + coalesced non-temporal store of attention ----
  float* attnBase = attn + ((size_t)b * SEQ + (size_t)qt * 16) * SEQ;
  for (int m = 0; m < 16; ++m) {
    const float inv = sRowInv[m];
    for (int col = t; col < SEQ; col += 128) {
      float p = sS[m * SSTRIDE + col] * inv;
      sS[m * SSTRIDE + col] = p;
      __builtin_nontemporal_store(p, &attnBase[(size_t)m * SEQ + col]);
    }
  }
  __syncthreads();

  // ---- Pass 2: context = P V; wave w owns d-columns [w*32, w*32+32) ----
  const v16bf* vpv = (const v16bf*)vp;
  v8f acc0 = {}, acc1 = {};
  for (int kc = 0; kc < SEQ / 32; ++kc) {
    v16bf ph, pl;
#pragma unroll
    for (int i = 0; i < 16; ++i) {
      const int Kl = i + (hg ? 16 : 0);
      float x = sS[lm * SSTRIDE + kc * 32 + Kl];
      __bf16 h, l; split_bf16(x, h, l);
      ph[i] = h; pl[i] = l;
    }
#pragma unroll
    for (int t2 = 0; t2 < 2; ++t2) {
      v16bf vh, vl;
      if (PACKED) {
        const size_t base = ((size_t)(((b * 64 + kc) * 8 + (w * 2 + t2)) * 32 + L)) * 2;
        vh = vpv[base]; vl = vpv[base + 1];
      } else {
#pragma unroll
        for (int i = 0; i < 16; ++i) {
          const int Kd = i + (hg ? 16 : 0);
          float x = V[(kc * 32 + Kd) * DIM + w * 32 + t2 * 16 + lm];
          __bf16 h, l; split_bf16(x, h, l);
          vh[i] = h; vl[i] = l;
        }
      }
      v8f& acc = t2 ? acc1 : acc0;
      acc = wmma_bf16(ph, vh, acc);
      acc = wmma_bf16(ph, vl, acc);
      acc = wmma_bf16(pl, vh, acc);
    }
  }

  float* ctxBase = ctx + batchBase + (size_t)qt * 16 * DIM;
#pragma unroll
  for (int t2 = 0; t2 < 2; ++t2) {
    v8f acc = t2 ? acc1 : acc0;
#pragma unroll
    for (int rr = 0; rr < 8; ++rr) {
      const int row = rr + (hg ? 8 : 0);
      __builtin_nontemporal_store(acc[rr], &ctxBase[row * DIM + w * 32 + t2 * 16 + lm]);
    }
  }
}

extern "C" void kernel_launch(void* const* d_in, const int* in_sizes, int n_in,
                              void* d_out, int out_size, void* d_ws, size_t ws_size,
                              hipStream_t stream) {
  (void)in_sizes; (void)n_in; (void)out_size;
  const float* q = (const float*)d_in[0];
  const float* k = (const float*)d_in[1];
  const float* v = (const float*)d_in[2];

  float* ctx  = (float*)d_out;                           // [32,2048,128]
  float* attn = (float*)d_out + (size_t)32 * SEQ * DIM;  // [32,2048,2048]

  const size_t smemBytes = (size_t)(16 * SSTRIDE + 128 + 16 + 16) * sizeof(float);
  const size_t packBytes = (size_t)(KP_ELEMS + VP_ELEMS) * sizeof(__bf16);  // 64 MiB

  dim3 grid(SEQ / 16, 32);
  dim3 block(128);

  if (d_ws != nullptr && ws_size >= packBytes) {
    __bf16* kp = (__bf16*)d_ws;
    __bf16* vp = kp + KP_ELEMS;
    // One-shot conversion passes (stream-ordered before the main kernel).
    pack_k_kernel<<<KP_ELEMS / 32 / 256, 256, 0, stream>>>(k, kp);
    pack_v_kernel<<<VP_ELEMS / 32 / 256, 256, 0, stream>>>(v, vp);
    (void)hipFuncSetAttribute((const void*)sdpa_kernel<true>,
                              hipFuncAttributeMaxDynamicSharedMemorySize, (int)smemBytes);
    sdpa_kernel<true><<<grid, block, smemBytes, stream>>>(q, k, v, kp, vp, ctx, attn);
  } else {
    (void)hipFuncSetAttribute((const void*)sdpa_kernel<false>,
                              hipFuncAttributeMaxDynamicSharedMemorySize, (int)smemBytes);
    sdpa_kernel<false><<<grid, block, smemBytes, stream>>>(q, k, v, nullptr, nullptr, ctx, attn);
  }
}
